// FuzzyAttention_9818295238776
// MI455X (gfx1250) — compile-verified, matrix operand order checked by reading, not run on hardware
//
#include <hip/hip_runtime.h>

// ---------------- types ----------------
typedef __attribute__((ext_vector_type(16))) __bf16 v16bf;
typedef __attribute__((ext_vector_type(8)))  __bf16 v8bf;
typedef __attribute__((ext_vector_type(8)))  float  v8f;
typedef __attribute__((ext_vector_type(4)))  int    v4i;

#define AS1 __attribute__((address_space(1)))
#define AS3 __attribute__((address_space(3)))

// ---------------- CDNA5 async global->LDS path (guarded) ----------------
#if defined(__has_builtin)
#if __has_builtin(__builtin_amdgcn_global_load_async_to_lds_b128)
#define HAVE_ASYNC_LDS 1
#endif
#endif

#ifndef HAVE_ASYNC_LDS
#define HAVE_ASYNC_LDS 0
#endif

#if HAVE_ASYNC_LDS
#if defined(__has_builtin) && __has_builtin(__builtin_amdgcn_s_wait_asynccnt)
#define WAIT_ASYNC(n) __builtin_amdgcn_s_wait_asynccnt(n)
#else
#define WAIT_ASYNC(n) asm volatile("s_wait_asynccnt %0" ::"i"(n) : "memory")
#endif
#else
#define WAIT_ASYNC(n)
#endif

// ---------------- problem sizes ----------------
#define B_SZ   16
#define N_SEQ  2048
#define D_DIM  1024
#define M_TOT  (B_SZ * N_SEQ)          // 32768 tokens
#define NX     ((size_t)M_TOT * D_DIM) // 33,554,432 elements of x
#define D2     ((size_t)D_DIM * D_DIM) // 1,048,576 weight elements

// ---------------- workspace layout (bytes) ----------------
#define OFF_XB   ((size_t)0)                       // bf16 x       : 64 MB
#define OFF_WQ   ((size_t)67108864)                // bf16 Wq      : 2 MB
#define OFF_WK   ((size_t)69206016)                // bf16 Wk      : 2 MB
#define OFF_WV   ((size_t)71303168)                // bf16 Wv      : 2 MB
#define OFF_CTX  ((size_t)73400320)                // f32 ctx      : 64 KB
#define OFF_QG   ((size_t)73465856)                // bf16 Qg      : 64 MB
#define OFF_K    ((size_t)140574720)               // bf16 K       : 64 MB
#define OFF_V    ((size_t)207683584)               // bf16 V       : 64 MB
#define OFF_W    ((size_t)274792448)               // f32 colw     : 128 KB

// ---------------- fragment helpers ----------------
// 16-bit A/B fragment (16x32 / 32x16), symmetric layout:
// lane l<16 : row l,    inner-dim cols {c0..c0+7, c0+16..c0+23}
// lane l>=16: row l-16, inner-dim cols {c0+8..c0+15, c0+24..c0+31}
__device__ __forceinline__ v16bf ld_frag(const __bf16* p0, const __bf16* p1) {
    union { v16bf v; v8bf h[2]; } u;
    u.h[0] = *(const v8bf*)p0;
    u.h[1] = *(const v8bf*)p1;
    return u.v;
}

__device__ __forceinline__ v8f wmma_bf16(v16bf a, v16bf b, v8f c) {
    return __builtin_amdgcn_wmma_f32_16x16x32_bf16(
        /*neg_a=*/false, a, /*neg_b=*/false, b,
        /*c_mod=*/(short)0, c, /*reuse_a=*/false, /*reuse_b=*/false);
}

// ---------------- kernel 0: fp32 -> bf16 convert ----------------
__global__ void k_convert(const float* __restrict__ in, __bf16* __restrict__ out, size_t n) {
    size_t stride = (size_t)gridDim.x * blockDim.x;
    for (size_t i = (size_t)blockIdx.x * blockDim.x + threadIdx.x; i < n; i += stride)
        out[i] = (__bf16)in[i];
}

// ---------------- kernel 0b: zero f32 buffer ----------------
__global__ void k_zero(float* __restrict__ p, size_t n) {
    size_t stride = (size_t)gridDim.x * blockDim.x;
    for (size_t i = (size_t)blockIdx.x * blockDim.x + threadIdx.x; i < n; i += stride)
        p[i] = 0.0f;
}

// ---------------- kernel 1: ctx = fc @ Wc^T (fp32, wave per output) ----------------
__global__ __launch_bounds__(256) void k_ctx(const float* __restrict__ fc,
                                             const float* __restrict__ Wc,
                                             float* __restrict__ ctx) {
    const int wave = threadIdx.x >> 5, lane = threadIdx.x & 31;
    const int idx  = blockIdx.x * 8 + wave;     // 0 .. 16383
    const int b = idx >> 10, e = idx & 1023;
    float acc = 0.0f;
    for (int d = lane; d < D_DIM; d += 32)
        acc += fc[b * D_DIM + d] * Wc[e * D_DIM + d];
#pragma unroll
    for (int o = 16; o; o >>= 1) acc += __shfl_xor(acc, o, 32);
    if (lane == 0) ctx[idx] = acc;
}

// ---------------- kernel 2: fused QKV projection GEMM (bf16 WMMA) ----------------
// z = 0: Qg = x@Wq^T + bq + ctx[b] ; z = 1: K = x@Wk^T + bk ; z = 2: V = x@Wv^T + bv
// Double-buffered LDS; tiles staged with GLOBAL_LOAD_ASYNC_TO_LDS_B128 when available.
#define TM 128
#define TN 128
#define TK 32
#define LDP (TK + 8)   // padded LDS row stride (bf16 elements)

__global__ __launch_bounds__(256) void k_qkv(
    const __bf16* __restrict__ X,
    const __bf16* __restrict__ Wq, const __bf16* __restrict__ Wk, const __bf16* __restrict__ Wv,
    const float*  __restrict__ bq, const float*  __restrict__ bk, const float*  __restrict__ bv,
    const float*  __restrict__ ctx,
    __bf16* __restrict__ Qg, __bf16* __restrict__ Ko, __bf16* __restrict__ Vo)
{
    const int which = blockIdx.z;
    const __bf16* W    = (which == 0) ? Wq : ((which == 1) ? Wk : Wv);
    const float*  bias = (which == 0) ? bq : ((which == 1) ? bk : bv);
    __bf16*       Out  = (which == 0) ? Qg : ((which == 1) ? Ko : Vo);

    __shared__ __bf16 As[2][TM][LDP];   // 2 x 10 KB
    __shared__ __bf16 Bs[2][TN][LDP];   // 2 x 10 KB

    const int tid  = threadIdx.x;
    const int wave = tid >> 5, lane = tid & 31;
    const int half = lane >> 4, lrow = lane & 15;
    const int m0 = blockIdx.y * TM;
    const int n0 = blockIdx.x * TN;
    const int wm = (wave & 3) * 32;   // wave's 32-row M slice
    const int wn = (wave >> 2) * 64;  // wave's 64-col N slice

    v8f acc[2][4];
#pragma unroll
    for (int i = 0; i < 2; i++)
#pragma unroll
        for (int j = 0; j < 4; j++) acc[i][j] = (v8f)(0.0f);

    // stage one 128x32 bf16 tile pair into LDS buffer s (4 x b128 per thread)
    auto issue = [&](int s, int k0) {
#pragma unroll
        for (int c = 0; c < 2; c++) {
            int chunk = tid + c * 256;        // 0..511
            int row   = chunk >> 2;
            int col8  = (chunk & 3) * 8;
            const __bf16* gA = &X[(size_t)(m0 + row) * D_DIM + k0 + col8];
            const __bf16* gB = &W[(size_t)(n0 + row) * D_DIM + k0 + col8];
#if HAVE_ASYNC_LDS
            __builtin_amdgcn_global_load_async_to_lds_b128(
                (AS1 v4i*)gA, (AS3 v4i*)&As[s][row][col8], 0, 0);
            __builtin_amdgcn_global_load_async_to_lds_b128(
                (AS1 v4i*)gB, (AS3 v4i*)&Bs[s][row][col8], 0, 0);
#else
            *(v8bf*)&As[s][row][col8] = *(const v8bf*)gA;
            *(v8bf*)&Bs[s][row][col8] = *(const v8bf*)gB;
            __builtin_prefetch(gA + TK, 0, 1);
            __builtin_prefetch(gB + TK, 0, 1);
#endif
        }
    };

    const int nT = D_DIM / TK;   // 32 k-tiles
    issue(0, 0);

    for (int t = 0; t < nT; t++) {
        const int cur = t & 1;
        if (t + 1 < nT) {
            issue(cur ^ 1, (t + 1) * TK);   // prefetch next tile into other buffer
            WAIT_ASYNC(4);                  // tile t landed (async loads complete in order)
        } else {
            WAIT_ASYNC(0);
        }
        __syncthreads();

        v16bf af[2], bfr[4];
#pragma unroll
        for (int i = 0; i < 2; i++) {
            int r = wm + i * 16 + lrow;
            af[i] = ld_frag(&As[cur][r][half * 8], &As[cur][r][16 + half * 8]);
        }
#pragma unroll
        for (int j = 0; j < 4; j++) {
            int r = wn + j * 16 + lrow;
            bfr[j] = ld_frag(&Bs[cur][r][half * 8], &Bs[cur][r][16 + half * 8]);
        }
#pragma unroll
        for (int i = 0; i < 2; i++)
#pragma unroll
            for (int j = 0; j < 4; j++)
                acc[i][j] = wmma_bf16(af[i], bfr[j], acc[i][j]);
        __syncthreads();   // all waves done reading buf[cur] before it is re-staged
    }

    // epilogue: bias (+ctx for Q), store bf16
    const int b = m0 >> 11;   // token block -> batch (128 | 2048)
#pragma unroll
    for (int i = 0; i < 2; i++)
#pragma unroll
        for (int j = 0; j < 4; j++) {
            int col = n0 + wn + j * 16 + lrow;
            float badd = bias[col] + ((which == 0) ? ctx[b * D_DIM + col] : 0.0f);
#pragma unroll
            for (int e = 0; e < 8; e++) {
                int row = m0 + wm + i * 16 + e + half * 8;
                Out[(size_t)row * D_DIM + col] = (__bf16)(acc[i][j][e] + badd);
            }
        }
}

// ---------------- kernel 3: scores + softmax + column weights ----------------
// grid: (N/16 query blocks, B). Block computes S[16][2048] in LDS via WMMA,
// softmax stats per row, then column sums atomically added into w[b][k].
#define QB 16

__global__ __launch_bounds__(256) void k_scores(
    const __bf16* __restrict__ Qg, const __bf16* __restrict__ Ko,
    float* __restrict__ w)
{
    extern __shared__ float S[];                 // [16][2048] = 128 KB
    __shared__ float rm[QB], rinv[QB];

    const int b  = blockIdx.y;
    const int q0 = blockIdx.x * QB;
    const int tid = threadIdx.x;
    const int wave = tid >> 5, lane = tid & 31;
    const int half = lane >> 4, lrow = lane & 15;
    const size_t base = (size_t)b * N_SEQ * D_DIM;
    const __bf16* Qp = Qg + base + (size_t)q0 * D_DIM;
    const __bf16* Kp = Ko + base;
    const float scale = 0.03125f;                // 1/sqrt(1024)

    // each wave owns 256 keys, processed as 4 chunks of 4 WMMA N-tiles
    for (int c = 0; c < 4; c++) {
        const int kb = wave * 256 + c * 64;
        v8f acc[4];
#pragma unroll
        for (int j = 0; j < 4; j++) acc[j] = (v8f)(0.0f);

        for (int k0 = 0; k0 < D_DIM; k0 += 32) {
            const __bf16* qr = Qp + (size_t)lrow * D_DIM + k0;
            v16bf af = ld_frag(qr + half * 8, qr + 16 + half * 8);
#pragma unroll
            for (int j = 0; j < 4; j++) {
                const __bf16* kr = Kp + (size_t)(kb + j * 16 + lrow) * D_DIM + k0;
                v16bf bfr = ld_frag(kr + half * 8, kr + 16 + half * 8);
                acc[j] = wmma_bf16(af, bfr, acc[j]);
            }
        }
#pragma unroll
        for (int j = 0; j < 4; j++)
#pragma unroll
            for (int e = 0; e < 8; e++)
                S[(e + 8 * half) * N_SEQ + kb + j * 16 + lrow] = acc[j][e] * scale;
    }
    __syncthreads();

    // per-row softmax stats: 2 rows per wave
#pragma unroll
    for (int r = 0; r < 2; r++) {
        const int row = wave * 2 + r;
        float mx = -3.4e38f;
        for (int j = lane; j < N_SEQ; j += 32) mx = fmaxf(mx, S[row * N_SEQ + j]);
#pragma unroll
        for (int o = 16; o; o >>= 1) mx = fmaxf(mx, __shfl_xor(mx, o, 32));
        float sm = 0.0f;
        for (int j = lane; j < N_SEQ; j += 32) sm += __expf(S[row * N_SEQ + j] - mx);
#pragma unroll
        for (int o = 16; o; o >>= 1) sm += __shfl_xor(sm, o, 32);
        if (lane == 0) { rm[row] = mx; rinv[row] = 1.0f / sm; }
    }
    __syncthreads();

    // column sums of normalized attention -> global atomic add
    for (int col = tid; col < N_SEQ; col += 256) {
        float s = 0.0f;
#pragma unroll
        for (int q = 0; q < QB; q++)
            s += __expf(S[q * N_SEQ + col] - rm[q]) * rinv[q];
        atomicAdd(&w[b * N_SEQ + col], s);
    }
}

// ---------------- kernel 4: out[b,d] = (1/N) * sum_k w[b,k] * V[b,k,d] ----------------
__global__ __launch_bounds__(256) void k_out(
    const float* __restrict__ w, const __bf16* __restrict__ Vo, float* __restrict__ out)
{
    const int b = blockIdx.x;
    const int tid = threadIdx.x;
    const __bf16* Vp = Vo + (size_t)b * N_SEQ * D_DIM;
    float acc[4] = {0.f, 0.f, 0.f, 0.f};
    for (int k = 0; k < N_SEQ; k++) {
        float wk = w[b * N_SEQ + k];
#pragma unroll
        for (int c = 0; c < 4; c++)
            acc[c] += wk * (float)Vp[(size_t)k * D_DIM + tid + c * 256];
    }
#pragma unroll
    for (int c = 0; c < 4; c++)
        out[b * D_DIM + tid + c * 256] = acc[c] * (1.0f / (float)N_SEQ);
}

// ---------------- launcher ----------------
extern "C" void kernel_launch(void* const* d_in, const int* in_sizes, int n_in,
                              void* d_out, int out_size, void* d_ws, size_t ws_size,
                              hipStream_t stream) {
    (void)in_sizes; (void)n_in; (void)out_size; (void)ws_size;
    const float* x  = (const float*)d_in[0];
    const float* fc = (const float*)d_in[1];
    const float* Wq = (const float*)d_in[2];
    const float* bq = (const float*)d_in[3];
    const float* Wk = (const float*)d_in[4];
    const float* bk = (const float*)d_in[5];
    const float* Wv = (const float*)d_in[6];
    const float* bv = (const float*)d_in[7];
    const float* Wc = (const float*)d_in[8];
    float* out = (float*)d_out;

    char* ws = (char*)d_ws;
    __bf16* xb  = (__bf16*)(ws + OFF_XB);
    __bf16* wqb = (__bf16*)(ws + OFF_WQ);
    __bf16* wkb = (__bf16*)(ws + OFF_WK);
    __bf16* wvb = (__bf16*)(ws + OFF_WV);
    float*  ctx = (float*) (ws + OFF_CTX);
    __bf16* Qgb = (__bf16*)(ws + OFF_QG);
    __bf16* Kb  = (__bf16*)(ws + OFF_K);
    __bf16* Vb  = (__bf16*)(ws + OFF_V);
    float*  wcw = (float*) (ws + OFF_W);

    // stage 0: precision staging + zeroing
    k_convert<<<4096, 256, 0, stream>>>(x,  xb,  NX);
    k_convert<<<1024, 256, 0, stream>>>(Wq, wqb, D2);
    k_convert<<<1024, 256, 0, stream>>>(Wk, wkb, D2);
    k_convert<<<1024, 256, 0, stream>>>(Wv, wvb, D2);
    k_zero<<<128, 256, 0, stream>>>(wcw, (size_t)B_SZ * N_SEQ);

    // stage 1: fuzzy context (fp32)
    k_ctx<<<2048, 256, 0, stream>>>(fc, Wc, ctx);

    // stage 2: fused QKV projections (WMMA bf16, async double-buffered LDS)
    k_qkv<<<dim3(D_DIM / TN, M_TOT / TM, 3), 256, 0, stream>>>(
        xb, wqb, wkb, wvb, bq, bk, bv, ctx, Qgb, Kb, Vb);

    // stage 3: scores + softmax + attention column weights (WMMA bf16 + LDS)
    k_scores<<<dim3(N_SEQ / QB, B_SZ), 256, QB * N_SEQ * sizeof(float), stream>>>(
        Qgb, Kb, wcw);

    // stage 4: weighted mean over V
    k_out<<<B_SZ, 256, 0, stream>>>(wcw, Vb, out);
}
